// Glom_10342281248996
// MI455X (gfx1250) — matrix operand order repaired and checked by reading, not compiled, not run on hardware
//
#include <hip/hip_runtime.h>

#define TKN   2048          // 8 batches * 256 tokens
#define DM    512
#define NLVL  6
#define NITER 12

typedef __bf16 bf16;
typedef __attribute__((ext_vector_type(16))) bf16  v16bf;
typedef __attribute__((ext_vector_type(8)))  bf16  v8bf;
typedef __attribute__((ext_vector_type(8)))  float v8f;

#define XSTR 520            // 512 + 8 bf16 pad (row stride -> 4-bank rotation)
#define CSTR 516            // 512 + 4 f32 pad

__device__ __forceinline__ v8f wmma_bf16(v16bf a, v16bf b, v8f c) {
  // D = A(16x32) * B(32x16) + C, f32 accumulate
  return __builtin_amdgcn_wmma_f32_16x16x32_bf16(false, a, false, b, (short)0, c,
                                                 false, false);
}

// A-operand (16x32 bf16) per-lane gather from a row-major LDS tile.
// ISA layout: lane L: m = L&15, ah = L>>4; a[j<8] = A[m][ah*8+j], a[j>=8] = A[m][16+ah*8+(j-8)]
__device__ __forceinline__ v16bf load_atile(const bf16* base, int stride, int mrow0,
                                            int k0, int lane) {
  const int m  = mrow0 + (lane & 15);
  const int ah = lane >> 4;
  const bf16* p = base + m * stride + k0 + ah * 8;
  v8bf lo = *(const v8bf*)p;
  v8bf hi = *(const v8bf*)(p + 16);
  v16bf a;
#pragma unroll
  for (int j = 0; j < 8; ++j) { a[j] = lo[j]; a[j + 8] = hi[j]; }
  return a;
}

__device__ __forceinline__ float gelu_exact(float x) {
  return 0.5f * x * (1.0f + erff(x * 0.70710678118654752f));
}

// ---------------------------------------------------------------------------
// Patch embedding: tokens[t][d] = patch(t) . W_patch[:,d] + b_patch[d]
// ---------------------------------------------------------------------------
__global__ __launch_bounds__(128) void patch_kernel(const float* __restrict__ img,
                                                    const float* __restrict__ Wp,
                                                    const float* __restrict__ bp,
                                                    float* __restrict__ tokens) {
  __shared__ float pv[588];
  const int t = blockIdx.x;                       // 0..2047
  const int d = blockIdx.y * 128 + threadIdx.x;   // 0..511
  const int b = t >> 8, p = t & 255, ph = p >> 4, pw = p & 15;
  for (int e = threadIdx.x; e < 588; e += 128) {
    int cch = e % 3, pc = (e / 3) % 14, pr = e / 42;
    pv[e] = img[(((size_t)b * 3 + cch) * 224 + (ph * 14 + pr)) * 224 + (pw * 14 + pc)];
  }
  __syncthreads();
  float acc = bp[d];
  for (int e = 0; e < 588; ++e) acc += pv[e] * Wp[(size_t)e * DM + d];
  tokens[(size_t)t * DM + d] = acc;
}

// ---------------------------------------------------------------------------
// Weight conversion: f32 row-major (K x N) -> bf16 WMMA-B tile layout.
// Tile (kt,nt) is 32x16; element j of lane L = W[kt*32 + (j>>3)*16 + (L>>4)*8 + (j&7)][nt*16 + (L&15)]
// stored contiguously: dst[(kt*NT+nt)*512 + L*16 + j]
// ---------------------------------------------------------------------------
__global__ __launch_bounds__(256) void convw_kernel(const float* __restrict__ src,
                                                    bf16* __restrict__ dst,
                                                    int K, int N) {
  int idx = blockIdx.x * 256 + threadIdx.x;
  if (idx >= K * N) return;
  int j = idx & 15, L = (idx >> 4) & 31;
  int rest = idx >> 9;
  int NT = N >> 4;
  int nt = rest % NT, kt = rest / NT;
  int k = kt * 32 + (j >> 3) * 16 + (L >> 4) * 8 + (j & 7);
  int n = nt * 16 + (L & 15);
  dst[idx] = (bf16)src[(size_t)k * N + n];
}

__global__ __launch_bounds__(256) void initlvl_kernel(const float* __restrict__ init,
                                                      float* __restrict__ lvl) {
  int idx = blockIdx.x * 256 + threadIdx.x;
  if (idx >= TKN * NLVL * DM) return;
  int d = idx & 511, l = (idx >> 9) % NLVL;
  lvl[idx] = init[l * DM + d];
}

// bf16 FF inputs: Xbu[l][t][:] = levels[t][l], Xtd[l][t][:] = levels[t][l+1] + pos[t%256]
__global__ __launch_bounds__(256) void prep_kernel(const float* __restrict__ lvl,
                                                   const float* __restrict__ pos,
                                                   bf16* __restrict__ Xbu,
                                                   bf16* __restrict__ Xtd) {
  int idx = blockIdx.x * 256 + threadIdx.x;
  if (idx >= 5 * TKN * DM) return;
  int d = idx & 511;
  int l = (idx >> 9) % 5;
  int t = idx / (512 * 5);
  size_t lbase = ((size_t)t * NLVL + l) * DM + d;
  size_t xo = ((size_t)l * TKN + t) * DM + d;
  Xbu[xo] = (bf16)lvl[lbase];
  Xtd[xo] = (bf16)(lvl[lbase + DM] + pos[(size_t)(t & 255) * DM + d]);
}

// norms[t*6+l] = ||levels[t][l][:]||, one wave per vector
__global__ __launch_bounds__(256) void norms_kernel(const float* __restrict__ lvl,
                                                    float* __restrict__ nrm) {
  int wid = blockIdx.x * 8 + (threadIdx.x >> 5);    // 0..12287
  int lane = threadIdx.x & 31;
  const float* p = lvl + (size_t)wid * DM;
  float s = 0.f;
  for (int k = lane; k < DM; k += 32) { float v = p[k]; s += v * v; }
#pragma unroll
  for (int off = 16; off > 0; off >>= 1) s += __shfl_xor(s, off, 32);
  if (lane == 0) nrm[wid] = sqrtf(s);
}

// ---------------------------------------------------------------------------
// Fused grouped FF: Y = gelu(X @ Win) @ Wout for one 32-row block, one level,
// one direction. Hidden 2048 processed in 4 chunks of 512; per chunk the bf16
// H tile lives in LDS; Y accumulates in v8f registers (2 m-tiles x 4 n-tiles
// per wave). B operands stream straight from the tiled bf16 weights (L2).
// ---------------------------------------------------------------------------
__global__ __launch_bounds__(256) void ff_kernel(
    const bf16* __restrict__ Xbu, const bf16* __restrict__ Xtd,
    const bf16* __restrict__ WbuIn, const bf16* __restrict__ WtdIn,
    const bf16* __restrict__ WbuOut, const bf16* __restrict__ WtdOut,
    float* __restrict__ Ybu, float* __restrict__ Ytd) {
  extern __shared__ char smem[];
  bf16* Xs = (bf16*)smem;                              // 32 x XSTR
  bf16* Hs = (bf16*)(smem + 32 * XSTR * sizeof(bf16)); // 32 x XSTR

  const int rb  = blockIdx.x;      // 0..63 : 32-row block
  const int l   = blockIdx.y;      // 0..4
  const int dir = blockIdx.z;      // 0 = bottom-up, 1 = top-down
  const bf16* X    = dir ? Xtd    : Xbu;
  const bf16* Win  = dir ? WtdIn  : WbuIn;
  const bf16* Wout = dir ? WtdOut : WbuOut;
  float* Y         = dir ? Ytd    : Ybu;

  const int tid = threadIdx.x;
  const int wave = tid >> 5, lane = tid & 31;
  const int row0 = rb * 32;

  // Stage X block (32 x 512 bf16) into LDS, 128B per thread, coalesced.
  {
    int r = tid >> 3;
    int c = (tid & 7) * 64;
    const uint4* g = (const uint4*)(X + ((size_t)l * TKN + row0 + r) * DM + c);
    uint4* s4 = (uint4*)(Xs + r * XSTR + c);
#pragma unroll
    for (int k = 0; k < 8; ++k) s4[k] = g[k];
  }
  __syncthreads();

  v8f acc[2][4];
#pragma unroll
  for (int mt = 0; mt < 2; ++mt)
#pragma unroll
    for (int q = 0; q < 4; ++q) acc[mt][q] = (v8f){0, 0, 0, 0, 0, 0, 0, 0};

  for (int c = 0; c < 4; ++c) {           // hidden chunks of 512
    // ---- stage 1: H = gelu(X @ Win[:, c*512 : c*512+512])
    v8f h[2][4];
#pragma unroll
    for (int mt = 0; mt < 2; ++mt)
#pragma unroll
      for (int q = 0; q < 4; ++q) h[mt][q] = (v8f){0, 0, 0, 0, 0, 0, 0, 0};

    for (int kk = 0; kk < 16; ++kk) {     // K over 512 input dims
      v16bf a0 = load_atile(Xs, XSTR, 0,  kk * 32, lane);
      v16bf a1 = load_atile(Xs, XSTR, 16, kk * 32, lane);
#pragma unroll
      for (int q = 0; q < 4; ++q) {
        int nt = c * 32 + wave * 4 + q;   // 0..127 hidden n-tile
        v16bf b = *(const v16bf*)(Win + (((size_t)l * 16 + kk) * 128 + nt) * 512 + lane * 16);
        h[0][q] = wmma_bf16(a0, b, h[0][q]);
        h[1][q] = wmma_bf16(a1, b, h[1][q]);
      }
    }
    // gelu -> bf16 H tile in LDS (D layout: lane L: n=L&15, m = mt*16 + (L>>4)*8 + v)
#pragma unroll
    for (int mt = 0; mt < 2; ++mt)
#pragma unroll
      for (int q = 0; q < 4; ++q) {
        int n = (wave * 4 + q) * 16 + (lane & 15);
        int mb = mt * 16 + (lane >> 4) * 8;
#pragma unroll
        for (int v = 0; v < 8; ++v)
          Hs[(mb + v) * XSTR + n] = (bf16)gelu_exact(h[mt][q][v]);
      }
    __syncthreads();

    // ---- stage 2: Y += H @ Wout[c*512 : c*512+512, :]
    for (int kk = 0; kk < 16; ++kk) {
      v16bf a0 = load_atile(Hs, XSTR, 0,  kk * 32, lane);
      v16bf a1 = load_atile(Hs, XSTR, 16, kk * 32, lane);
      int ktile = l * 64 + c * 16 + kk;   // flat 32-row tile index into Wout
#pragma unroll
      for (int q = 0; q < 4; ++q) {
        v16bf b = *(const v16bf*)(Wout + ((size_t)ktile * 32 + wave * 4 + q) * 512 + lane * 16);
        acc[0][q] = wmma_bf16(a0, b, acc[0][q]);
        acc[1][q] = wmma_bf16(a1, b, acc[1][q]);
      }
    }
    __syncthreads();
  }

  // Write Y (f32)
#pragma unroll
  for (int mt = 0; mt < 2; ++mt)
#pragma unroll
    for (int q = 0; q < 4; ++q) {
      int n = (wave * 4 + q) * 16 + (lane & 15);
      int mb = mt * 16 + (lane >> 4) * 8;
#pragma unroll
      for (int v = 0; v < 8; ++v)
        Y[((size_t)l * TKN + row0 + mb + v) * DM + n] = acc[mt][q][v];
    }
}

// ---------------------------------------------------------------------------
// Consensus attention + final combine for one (batch, level, 16-row i-block).
// sim[i][j] = (x_i . x_j) / max(||x_j||,1e-12) * d^-0.5 ; diag -> -0.0005 ;
// softmax over j ; cons = attn @ X ; out = 0.25*(lvl + bu + td + cons)
// ---------------------------------------------------------------------------
__global__ __launch_bounds__(256) void cons_kernel(
    const float* __restrict__ lvl, const float* __restrict__ nrm,
    const float* __restrict__ tokens, const float* __restrict__ Ybu,
    const float* __restrict__ Ytd, float* __restrict__ out) {
  extern __shared__ char smem[];
  float* Xi = (float*)smem;                              // 16 x CSTR
  float* Xj = (float*)(smem + 16 * CSTR * 4);            // 32 x CSTR
  float* S  = (float*)(smem + 48 * CSTR * 4);            // 16 x 256
  float* nr = (float*)(smem + 48 * CSTR * 4 + 16 * 256 * 4);  // 256

  const int ib = blockIdx.x;   // 0..15
  const int l  = blockIdx.y;   // 0..5
  const int b  = blockIdx.z;   // 0..7
  const int tid = threadIdx.x;
  const int i0 = ib * 16;
  const int t0 = b * 256;

  {  // load Xi (16 x 512)
    int r = tid >> 4, c = (tid & 15) * 32;
    const float4* g = (const float4*)(lvl + ((size_t)(t0 + i0 + r) * NLVL + l) * DM + c);
    float4* s4 = (float4*)(Xi + r * CSTR + c);
#pragma unroll
    for (int k = 0; k < 8; ++k) s4[k] = g[k];
  }
  nr[tid] = nrm[(size_t)(t0 + tid) * NLVL + l];
  __syncthreads();

  // ---- Gram: S[ii][j] = x_(i0+ii) . x_j
  for (int jt = 0; jt < 8; ++jt) {
    int j0 = jt * 32;
    {
      int r = tid >> 3, c = (tid & 7) * 64;
      const float4* g = (const float4*)(lvl + ((size_t)(t0 + j0 + r) * NLVL + l) * DM + c);
      float4* s4 = (float4*)(Xj + r * CSTR + c);
#pragma unroll
      for (int k = 0; k < 16; ++k) s4[k] = g[k];
    }
    __syncthreads();
    int jj = tid & 31;
#pragma unroll
    for (int rep = 0; rep < 2; ++rep) {
      int ii = (tid >> 5) * 2 + rep;
      const float* xi = Xi + ii * CSTR;
      const float* xj = Xj + jj * CSTR;
      float dt = 0.f;
#pragma unroll 8
      for (int k = 0; k < DM; ++k) dt += xi[k] * xj[k];
      S[ii * 256 + j0 + jj] = dt;
    }
    __syncthreads();
  }

  // ---- scale + diag + softmax (one wave handles 2 rows)
  {
    int wave = tid >> 5, lane = tid & 31;
#pragma unroll
    for (int rr = 0; rr < 2; ++rr) {
      int row = wave * 2 + rr;
      float v[8];
#pragma unroll
      for (int k = 0; k < 8; ++k) {
        int j = lane + k * 32;
        float s = S[row * 256 + j] / fmaxf(nr[j], 1e-12f) * 0.044194173824159216f;
        if (j == i0 + row) s = -0.0005f;
        v[k] = s;
      }
      float mx = v[0];
#pragma unroll
      for (int k = 1; k < 8; ++k) mx = fmaxf(mx, v[k]);
#pragma unroll
      for (int off = 16; off > 0; off >>= 1) mx = fmaxf(mx, __shfl_xor(mx, off, 32));
      float sum = 0.f;
#pragma unroll
      for (int k = 0; k < 8; ++k) { v[k] = expf(v[k] - mx); sum += v[k]; }
#pragma unroll
      for (int off = 16; off > 0; off >>= 1) sum += __shfl_xor(sum, off, 32);
      float inv = 1.0f / sum;
#pragma unroll
      for (int k = 0; k < 8; ++k) S[row * 256 + lane + k * 32] = v[k] * inv;
    }
  }
  __syncthreads();

  // ---- cons = attn @ X   (thread owns row ii, 8 interleaved float4 columns)
  float4 acc[8];
#pragma unroll
  for (int k = 0; k < 8; ++k) acc[k] = (float4){0, 0, 0, 0};
  const int ii = tid >> 4;
  const int dbase = (tid & 15) * 4;

  for (int jt = 0; jt < 8; ++jt) {
    int j0 = jt * 32;
    {
      int r = tid >> 3, c = (tid & 7) * 64;
      const float4* g = (const float4*)(lvl + ((size_t)(t0 + j0 + r) * NLVL + l) * DM + c);
      float4* s4 = (float4*)(Xj + r * CSTR + c);
#pragma unroll
      for (int k = 0; k < 16; ++k) s4[k] = g[k];
    }
    __syncthreads();
    for (int jj = 0; jj < 32; ++jj) {
      float a = S[ii * 256 + j0 + jj];
      const float* xr = Xj + jj * CSTR + dbase;
#pragma unroll
      for (int k = 0; k < 8; ++k) {
        float4 x = *(const float4*)(xr + k * 64);
        acc[k].x += a * x.x; acc[k].y += a * x.y;
        acc[k].z += a * x.z; acc[k].w += a * x.w;
      }
    }
    __syncthreads();
  }

  // ---- combine: out = 0.25 * (lvl + bu + td + cons)
  const int t = t0 + i0 + ii;
#pragma unroll
  for (int k = 0; k < 8; ++k) {
    int d = dbase + k * 64;
    float4 lv = *(const float4*)(Xi + ii * CSTR + d);
    float4 bu, td;
    if (l == 0) bu = *(const float4*)(tokens + (size_t)t * DM + d);
    else        bu = *(const float4*)(Ybu + ((size_t)(l - 1) * TKN + t) * DM + d);
    if (l < 5)  td = *(const float4*)(Ytd + ((size_t)l * TKN + t) * DM + d);
    else        td = (float4){0, 0, 0, 0};
    float4 o;
    o.x = 0.25f * (lv.x + bu.x + td.x + acc[k].x);
    o.y = 0.25f * (lv.y + bu.y + td.y + acc[k].y);
    o.z = 0.25f * (lv.z + bu.z + td.z + acc[k].z);
    o.w = 0.25f * (lv.w + bu.w + td.w + acc[k].w);
    *(float4*)(out + ((size_t)t * NLVL + l) * DM + d) = o;
  }
}

// ---------------------------------------------------------------------------
extern "C" void kernel_launch(void* const* d_in, const int* in_sizes, int n_in,
                              void* d_out, int out_size, void* d_ws, size_t ws_size,
                              hipStream_t stream) {
  (void)in_sizes; (void)n_in; (void)out_size; (void)ws_size;
  const float* img    = (const float*)d_in[0];
  const float* Wp     = (const float*)d_in[1];
  const float* bp     = (const float*)d_in[2];
  const float* pos    = (const float*)d_in[3];
  const float* initl  = (const float*)d_in[4];
  const float* bu_in  = (const float*)d_in[5];
  const float* bu_out = (const float*)d_in[6];
  const float* td_in  = (const float*)d_in[7];
  const float* td_out = (const float*)d_in[8];

  char* ws = (char*)d_ws;
  size_t off = 0;
  auto alloc = [&](size_t bytes) -> void* {
    void* p = ws + off;
    off = (off + bytes + 255) & ~(size_t)255;
    return p;
  };
  const size_t WINSZ  = 5ull * 512 * 2048;   // elements (both in/out weights)
  bf16*  WbuIn  = (bf16*)alloc(WINSZ * 2);
  bf16*  WtdIn  = (bf16*)alloc(WINSZ * 2);
  bf16*  WbuOut = (bf16*)alloc(WINSZ * 2);
  bf16*  WtdOut = (bf16*)alloc(WINSZ * 2);
  float* tokens = (float*)alloc((size_t)TKN * DM * 4);
  float* lvlA   = (float*)alloc((size_t)TKN * NLVL * DM * 4);
  float* lvlB   = (float*)alloc((size_t)TKN * NLVL * DM * 4);
  bf16*  Xbu    = (bf16*)alloc(5ull * TKN * DM * 2);
  bf16*  Xtd    = (bf16*)alloc(5ull * TKN * DM * 2);
  float* Ybu    = (float*)alloc(5ull * TKN * DM * 4);
  float* Ytd    = (float*)alloc(5ull * TKN * DM * 4);
  float* nrmB   = (float*)alloc((size_t)TKN * NLVL * 4);

  const int cvBlocks = (int)((WINSZ + 255) / 256);
  convw_kernel<<<cvBlocks, 256, 0, stream>>>(bu_in, WbuIn, 5 * 512, 2048);
  convw_kernel<<<cvBlocks, 256, 0, stream>>>(td_in, WtdIn, 5 * 512, 2048);
  convw_kernel<<<cvBlocks, 256, 0, stream>>>(bu_out, WbuOut, 5 * 2048, 512);
  convw_kernel<<<cvBlocks, 256, 0, stream>>>(td_out, WtdOut, 5 * 2048, 512);
  patch_kernel<<<dim3(TKN, 4), 128, 0, stream>>>(img, Wp, bp, tokens);
  initlvl_kernel<<<(TKN * NLVL * DM) / 256, 256, 0, stream>>>(initl, lvlA);

  const size_t ffLds   = 2 * 32 * XSTR * sizeof(bf16);                  // 66,560 B
  const size_t consLds = 48 * CSTR * 4 + 16 * 256 * 4 + 256 * 4;        // 116,480 B

  float* cur = lvlA;
  float* nxt = lvlB;
  for (int it = 0; it < NITER; ++it) {
    prep_kernel<<<(5 * TKN * DM) / 256, 256, 0, stream>>>(cur, pos, Xbu, Xtd);
    norms_kernel<<<(TKN * NLVL) / 8, 256, 0, stream>>>(cur, nrmB);
    ff_kernel<<<dim3(64, 5, 2), 256, ffLds, stream>>>(Xbu, Xtd, WbuIn, WtdIn,
                                                      WbuOut, WtdOut, Ybu, Ytd);
    float* outp = (it == NITER - 1) ? (float*)d_out : nxt;
    cons_kernel<<<dim3(16, NLVL, 8), 256, consLds, stream>>>(cur, nrmB, tokens,
                                                             Ybu, Ytd, outp);
    float* tmp = cur; cur = nxt; nxt = tmp;
  }
}
